// ProteinFeatures_60017872994848
// MI455X (gfx1250) — matrix operand order repaired and checked by reading, not compiled, not run on hardware
//
#include <hip/hip_runtime.h>
#include <hip/hip_bf16.h>
#include <math.h>

// ProteinFeatures for MI455X (gfx1250, wave32).
// Fixed problem shape from setup_inputs(): B=2, L=2048, K=48, 14 atoms, 416-dim
// edge features, 128-dim output. Slot tables are constant => S is unused.

typedef __attribute__((ext_vector_type(16))) _Float16 v16h;
typedef __attribute__((ext_vector_type(8)))  float    v8f;

#define BATCH 2
#define SEQ   2048
#define KNN   48
#define NATM  14
#define FDIM  416     // 16 pos + 25*16 rbf
#define EDIM  128
#define KCHUNKS 13    // 416 / 32

// ---------------------------------------------------------------------------
// Kernel 0: pre-swizzle edge_W (f32 416x128) into per-lane WMMA B-fragment
// order, f16. Layout: [kb][ntile][lane][16 halves] -> 32 contiguous bytes per
// lane, so the GEMM inner loop is one coalesced 32B load per lane per K-step.
// B order used here (contiguous k, not AMD's table order): lanes 0-15 hold K=0..15, lanes 16-31 K=16..31,
// column N = lane%16, VGPR v packs K=2v,2v+1 (halves h -> K = h + 16*(lane/16)).
// ---------------------------------------------------------------------------
__global__ void pf_cvtw(const float* __restrict__ edgeW, _Float16* __restrict__ wfrag) {
    int idx = blockIdx.x * blockDim.x + threadIdx.x;           // 13*8*32*16
    if (idx >= KCHUNKS * 8 * 32 * 16) return;
    int h    = idx & 15;
    int lane = (idx >> 4) & 31;
    int nt   = (idx >> 9) & 7;
    int kb   = idx >> 12;
    int hiH  = lane >> 4;
    int krow = kb * 32 + h + hiH * 16;
    int col  = nt * 16 + (lane & 15);
    wfrag[idx] = (_Float16)edgeW[(size_t)krow * EDIM + col];
}

// ---------------------------------------------------------------------------
// Kernel 1: build X_all (5 atoms/token: bb0..3 + virtual) and the mask.
// Representative atom (Xc) == X_all atom 1 (backbone slot 1).
// ---------------------------------------------------------------------------
__global__ void pf_prep(const float* __restrict__ X, const float* __restrict__ Xm,
                        float* __restrict__ xall, float* __restrict__ msk) {
    int bi = blockIdx.x * blockDim.x + threadIdx.x;
    if (bi >= BATCH * SEQ) return;
    const float* xa = X + (size_t)bi * NATM * 3;
    float c0[3], x1[3], x2[3], b1[3], b2[3], nr[3];
    #pragma unroll
    for (int c = 0; c < 3; ++c) { c0[c] = xa[3 + c]; x1[c] = xa[c]; x2[c] = xa[6 + c]; }
    #pragma unroll
    for (int c = 0; c < 3; ++c) { b1[c] = x1[c] - c0[c]; b2[c] = x2[c] - c0[c]; }
    nr[0] = b1[1]*b2[2] - b1[2]*b2[1];
    nr[1] = b1[2]*b2[0] - b1[0]*b2[2];
    nr[2] = b1[0]*b2[1] - b1[1]*b2[0];
    float* o = xall + (size_t)bi * 15;
    #pragma unroll
    for (int a = 0; a < 4; ++a)
        #pragma unroll
        for (int c = 0; c < 3; ++c) o[a*3 + c] = xa[a*3 + c];
    #pragma unroll
    for (int c = 0; c < 3; ++c)
        o[12 + c] = 0.58273431f*nr[c] - 0.56802827f*b1[c] - 0.54067466f*b2[c] + c0[c];
    msk[bi] = Xm[(size_t)bi * NATM + 1];   // BB slot 1 mask
}

// ---------------------------------------------------------------------------
// Kernel 2: kNN. One workgroup per (b,i). Distances in LDS, then 48 parallel
// argmin passes (ascending distance, ties -> lowest index, matching top_k).
// ---------------------------------------------------------------------------
__global__ __launch_bounds__(256) void pf_knn(const float* __restrict__ xall,
                                              const float* __restrict__ msk,
                                              int* __restrict__ eidx,
                                              float* __restrict__ eidx_f) {
    __shared__ float Dadj[SEQ];
    __shared__ float sv[256];
    __shared__ int   si[256];
    const int bi = blockIdx.x;
    const int bb = bi / SEQ;
    const int t  = threadIdx.x;
    const int jbase = bb * SEQ;

    const float* xi = xall + (size_t)bi * 15 + 3;   // rep atom = atom 1
    const float xix = xi[0], xiy = xi[1], xiz = xi[2];
    const float mi  = msk[bi];

    for (int j = t; j < SEQ; j += 256) {
        const float* xj = xall + (size_t)(jbase + j) * 15 + 3;
        float dx = xix - xj[0], dy = xiy - xj[1], dz = xiz - xj[2];
        float d  = sqrtf(dx*dx + dy*dy + dz*dz + 1e-6f);
        Dadj[j] = d + (1.0f - mi * msk[jbase + j]) * 1e6f;
    }
    __syncthreads();

    for (int kk = 0; kk < KNN; ++kk) {
        float best = 3.0e38f; int bidx = SEQ;
        for (int j = t; j < SEQ; j += 256) {
            float v = Dadj[j];
            if (v < best) { best = v; bidx = j; }   // strided scan keeps lowest j on ties
        }
        sv[t] = best; si[t] = bidx;
        __syncthreads();
        for (int s = 128; s > 0; s >>= 1) {
            if (t < s) {
                float ov = sv[t + s]; int oi = si[t + s];
                if (ov < sv[t] || (ov == sv[t] && oi < si[t])) { sv[t] = ov; si[t] = oi; }
            }
            __syncthreads();
        }
        if (t == 0) {
            int bj = si[0];
            eidx  [(size_t)bi * KNN + kk] = bj;
            eidx_f[(size_t)bi * KNN + kk] = (float)bj;   // tuple output #2
            Dadj[bj] = 3.0e38f;
        }
        __syncthreads();
    }
}

// ---------------------------------------------------------------------------
// Kernel 3: fused edge features + WMMA GEMM (416x128) + LayerNorm.
// One workgroup = 16 edges (one M-tile). 8 waves, each owns one 16-wide N-tile
// and accumulates 13 chunks of K=32 with v_wmma_f32_16x16x32_f16 (unrolled).
// ---------------------------------------------------------------------------
__global__ __launch_bounds__(256) void pf_edge(const float* __restrict__ xall,
                                               const int*   __restrict__ eidx,
                                               const int*   __restrict__ ridx,
                                               const float* __restrict__ posW,
                                               const float* __restrict__ posb,
                                               const _Float16* __restrict__ wfrag,
                                               const float* __restrict__ gamma,
                                               const float* __restrict__ beta,
                                               float* __restrict__ Eout) {
    __shared__ float    Xi[15];
    __shared__ float    Xn[16][15];
    __shared__ int      jn[16];
    __shared__ int      dclip[16];
    __shared__ float    Dp[16][25];
    __shared__ _Float16 Fh[16][FDIM];     // 13.3 KB f16 feature tile (WMMA A)
    __shared__ float    Ef[16][EDIM];     // pre-LN output tile
    __shared__ float    rmean[16], rrstd[16];

    const int tile = blockIdx.x;
    const int bi   = tile / (KNN / 16);          // flattened (b,i)
    const int k0   = (tile % (KNN / 16)) * 16;   // neighbor sub-range
    const int bb   = bi / SEQ;
    const int t    = threadIdx.x;
    const int wave = t >> 5, lane = t & 31;

    // Warm up L1/L2 with this wave's pre-swizzled weight slice (global_prefetch_b8)
    const v16h* wfv = (const v16h*)wfrag;        // 32B-aligned fragments
    #pragma unroll
    for (int kb = 0; kb < KCHUNKS; ++kb)
        __builtin_prefetch((const void*)&wfv[(kb * 8 + wave) * 32 + lane], 0, 3);

    if (t < 16) {
        int j = eidx[(size_t)bi * KNN + k0 + t];
        jn[t] = j;
        int dc = (ridx[bi] - ridx[bb * SEQ + j]) + 32;
        dclip[t] = dc < 0 ? 0 : (dc > 64 ? 64 : dc);
    }
    __syncthreads();
    if (t < 240)      Xn[t / 15][t % 15] = xall[(size_t)(bb * SEQ + jn[t / 15]) * 15 + (t % 15)];
    else if (t < 255) Xi[t - 240]        = xall[(size_t)bi * 15 + (t - 240)];
    __syncthreads();

    // 25 atom-pair distances per edge
    for (int idx = t; idx < 16 * 25; idx += 256) {
        int e = idx / 25, p = idx % 25, ai = p / 5, aj = p % 5;
        float dx = Xi[ai*3+0] - Xn[e][aj*3+0];
        float dy = Xi[ai*3+1] - Xn[e][aj*3+1];
        float dz = Xi[ai*3+2] - Xn[e][aj*3+2];
        Dp[e][p] = sqrtf(dx*dx + dy*dy + dz*dz + 1e-6f);
    }
    __syncthreads();

    // 416-dim feature: [16 pos | 400 rbf], cast to f16 for WMMA A
    for (int idx = t; idx < 16 * FDIM; idx += 256) {
        int e = idx / FDIM, f = idx % FDIM;
        float v;
        if (f < 16) {
            v = posW[dclip[e] * 16 + f] + posb[f];
        } else {
            int q = f - 16, p = q >> 4, bin = q & 15;
            float mu = 2.0f + (20.0f / 15.0f) * (float)bin;   // linspace(2,22,16)
            float x  = (Dp[e][p] - mu) * 0.8f;                // / sigma(=1.25)
            v = __expf(-x * x);
        }
        Fh[e][f] = (_Float16)v;
    }
    __syncthreads();

    // WMMA: wave w -> output columns [w*16, w*16+16)
    const int n0  = wave * 16;
    const int m   = lane & 15;       // A row / C column within tile
    const int hiH = lane >> 4;       // half-wave selector
    v8f acc = {};
    #pragma unroll
    for (int kb = 0; kb < KCHUNKS; ++kb) {
        // A from LDS: same contiguous k order as B (not AMD's table order) -> one contiguous 32B chunk/lane (K = h + 16*hiH)
        v16h afr;
        #pragma unroll
        for (int h = 0; h < 16; ++h) {
            int kA = h + (hiH ? 16 : 0);
            afr[h] = Fh[m][kb * 32 + kA];
        }
        // B from pre-swizzled f16 weights: one 32B contiguous load per lane
        v16h bfr = wfv[(kb * 8 + wave) * 32 + lane];
        acc = __builtin_amdgcn_wmma_f32_16x16x32_f16(
                  false, afr, false, bfr, (short)0, acc, false, false);
    }
    #pragma unroll
    for (int r = 0; r < 8; ++r)           // C/D layout: VGPR r -> M = r + 8*hiH
        Ef[r + hiH * 8][n0 + m] = acc[r];
    __syncthreads();

    // LayerNorm over the 128 channels (population variance, eps 1e-5)
    if (t < 16) {
        float s = 0.f, ss = 0.f;
        for (int c = 0; c < EDIM; ++c) { float v = Ef[t][c]; s += v; ss += v * v; }
        float mean = s * (1.0f / EDIM);
        float var  = ss * (1.0f / EDIM) - mean * mean;
        rmean[t] = mean;
        rrstd[t] = rsqrtf(var + 1e-5f);
    }
    __syncthreads();
    for (int idx = t; idx < 16 * EDIM; idx += 256) {
        int r = idx / EDIM, c = idx % EDIM;
        float v = (Ef[r][c] - rmean[r]) * rrstd[r] * gamma[c] + beta[c];
        Eout[(size_t)(bi * KNN + k0 + r) * EDIM + c] = v;
    }
}

// ---------------------------------------------------------------------------
extern "C" void kernel_launch(void* const* d_in, const int* in_sizes, int n_in,
                              void* d_out, int out_size, void* d_ws, size_t ws_size,
                              hipStream_t stream) {
    const float* X     = (const float*)d_in[0];   // (B,L,14,3)
    const float* Xm    = (const float*)d_in[1];   // (B,L,14)
    // d_in[2] = S : unused (slot tables are constant)
    const int*   ridx  = (const int*)  d_in[3];   // (B,L)
    const float* posW  = (const float*)d_in[4];   // (65,16)
    const float* posb  = (const float*)d_in[5];   // (16,)
    const float* edgeW = (const float*)d_in[6];   // (416,128)
    const float* gma   = (const float*)d_in[7];   // (128,)
    const float* bta   = (const float*)d_in[8];   // (128,)

    float* out    = (float*)d_out;
    float* Eout   = out;                                          // (B,L,K,128)
    float* eidx_f = out + (size_t)BATCH * SEQ * KNN * EDIM;       // (B,L,K) as floats

    // Workspace: X_all (B*L*15 f32) | mask (B*L f32) | E_idx (B*L*K i32)
    //          | wfrag (13*8*32*16 f16, 32B-aligned at 1 MB offset)  ~= 1.10 MB
    float*    ws_f  = (float*)d_ws;
    float*    xall  = ws_f;
    float*    msk   = ws_f + (size_t)BATCH * SEQ * 15;
    int*      eidx  = (int*)(msk + BATCH * SEQ);
    _Float16* wfrag = (_Float16*)(eidx + (size_t)BATCH * SEQ * KNN);

    pf_cvtw<<<(KCHUNKS * 8 * 32 * 16 + 255) / 256, 256, 0, stream>>>(edgeW, wfrag);
    pf_prep<<<(BATCH * SEQ + 255) / 256, 256, 0, stream>>>(X, Xm, xall, msk);
    pf_knn <<<BATCH * SEQ, 256, 0, stream>>>(xall, msk, eidx, eidx_f);
    pf_edge<<<BATCH * SEQ * (KNN / 16), 256, 0, stream>>>(
        xall, eidx, ridx, posW, posb, wfrag, gma, bta, Eout);
}
